// ToVisualScatter_35253091565775
// MI455X (gfx1250) — compile-verified, matrix operand order checked by reading, not run on hardware
//
#include <hip/hip_runtime.h>

#define U_CNT 32768
#define F_IN  512
#define H1    512
#define H2    256
#define GY    64
#define GX    64
#define PAD_W 66   // 64 + 1 halo each side for 3x3 SAME conv

typedef __attribute__((ext_vector_type(16))) __bf16 v16bf;
typedef __attribute__((ext_vector_type(8)))  __bf16 v8bf;
typedef __attribute__((ext_vector_type(8)))  float  v8f;

union FragBF { v16bf v; v8bf h[2]; };

__device__ __forceinline__ unsigned short f2bf(float f) {
  unsigned u = __float_as_uint(f);
  u += 0x7FFFu + ((u >> 16) & 1u);   // round-to-nearest-even
  return (unsigned short)(u >> 16);
}

// Fragment loaders: per the ISA 16-bit 16x32 A layout, a lane's 16 bf16 live in
// two 16B chunks: [k .. k+7] and [k+16 .. k+23] (hi half-wave shifted by koff=8).
__device__ __forceinline__ void load_a2(FragBF a[2], const __bf16* arow0,
                                        const __bf16* arow1, int k) {
  a[0].h[0] = *(const v8bf*)(arow0 + k);
  a[0].h[1] = *(const v8bf*)(arow0 + k + 16);
  a[1].h[0] = *(const v8bf*)(arow1 + k);
  a[1].h[1] = *(const v8bf*)(arow1 + k + 16);
}

__device__ __forceinline__ void load_b4(FragBF b[4], const __bf16* brow,
                                        int k, int K) {
#pragma unroll
  for (int t = 0; t < 4; ++t) {
    const __bf16* p = brow + (size_t)t * 16 * K + k;
    b[t].h[0] = *(const v8bf*)(p);
    b[t].h[1] = *(const v8bf*)(p + 16);
  }
}

__device__ __forceinline__ void wmma_step(v8f acc[2][4], const FragBF a[2],
                                          const FragBF b[4]) {
#pragma unroll
  for (int mt = 0; mt < 2; ++mt)
#pragma unroll
    for (int t = 0; t < 4; ++t)
      acc[mt][t] = __builtin_amdgcn_wmma_f32_16x16x32_bf16(
          false, a[mt].v, false, b[t].v, (short)0, acc[mt][t], false, false);
}

// ---------------------------------------------------------------------------
// Row LayerNorm -> relu -> bf16, generic (out has same [row, C] layout)
// ---------------------------------------------------------------------------
__global__ __launch_bounds__(256) void ln_relu_bf16(
    const float* __restrict__ in, const float* __restrict__ sc,
    const float* __restrict__ bi, unsigned short* __restrict__ out, int C)
{
  __shared__ float ssum[256];
  __shared__ float ssq[256];
  const int row = blockIdx.x;
  const int tid = threadIdx.x;
  const float* x = in + (size_t)row * C;
  float s = 0.f, q = 0.f;
  for (int i = tid; i < C; i += 256) { float v = x[i]; s += v; q += v * v; }
  ssum[tid] = s; ssq[tid] = q; __syncthreads();
  for (int st = 128; st > 0; st >>= 1) {
    if (tid < st) { ssum[tid] += ssum[tid + st]; ssq[tid] += ssq[tid + st]; }
    __syncthreads();
  }
  const float invC = 1.0f / (float)C;
  const float mean = ssum[0] * invC;
  const float var  = fmaxf(ssq[0] * invC - mean * mean, 0.f);
  const float rstd = rsqrtf(var + 1e-6f);
  unsigned short* o = out + (size_t)row * C;
  for (int i = tid; i < C; i += 256) {
    float v = (x[i] - mean) * rstd * sc[i] + bi[i];
    o[i] = f2bf(fmaxf(v, 0.f));
  }
}

// ---------------------------------------------------------------------------
// LayerNorm -> relu -> bf16 into padded halo buffer Gp[66][66][256]
// ---------------------------------------------------------------------------
__global__ __launch_bounds__(256) void ln_relu_grid_pad(
    const float* __restrict__ grid, const float* __restrict__ sc,
    const float* __restrict__ bi, unsigned short* __restrict__ Gp)
{
  __shared__ float ssum[256];
  __shared__ float ssq[256];
  const int row = blockIdx.x;            // pixel index, 0..4095
  const int tid = threadIdx.x;
  const float* x = grid + (size_t)row * H2;
  float v0 = x[tid];
  ssum[tid] = v0; ssq[tid] = v0 * v0; __syncthreads();
  for (int st = 128; st > 0; st >>= 1) {
    if (tid < st) { ssum[tid] += ssum[tid + st]; ssq[tid] += ssq[tid + st]; }
    __syncthreads();
  }
  const float invC = 1.0f / (float)H2;
  const float mean = ssum[0] * invC;
  const float var  = fmaxf(ssq[0] * invC - mean * mean, 0.f);
  const float rstd = rsqrtf(var + 1e-6f);
  const int y = row >> 6, xc = row & 63;
  unsigned short* o = Gp + ((size_t)(y + 1) * PAD_W + (xc + 1)) * H2;
  float v = (v0 - mean) * rstd * sc[tid] + bi[tid];
  o[tid] = f2bf(fmaxf(v, 0.f));
}

// ---------------------------------------------------------------------------
// Weight conversions (f32 -> bf16, transposed so B-fragments load like A)
// ---------------------------------------------------------------------------
__global__ void cvt_transpose_bf16(const float* __restrict__ W,   // [K][N]
                                   unsigned short* __restrict__ Wt, // [N][K]
                                   int K, int N)
{
  int idx = blockIdx.x * blockDim.x + threadIdx.x;
  if (idx >= K * N) return;
  int k = idx / N, n = idx % N;
  Wt[(size_t)n * K + k] = f2bf(W[idx]);
}

__global__ void cvt_convw_bf16(const float* __restrict__ Win,   // [3][3][ci][co]
                               unsigned short* __restrict__ Wc) // [9][co][ci]
{
  int idx = blockIdx.x * blockDim.x + threadIdx.x;
  if (idx >= 9 * H2 * H2) return;
  int tap = idx / (H2 * H2);
  int rem = idx % (H2 * H2);
  int ci = rem / H2, co = rem % H2;
  Wc[(size_t)tap * H2 * H2 + (size_t)co * H2 + ci] = f2bf(Win[idx]);
}

// ---------------------------------------------------------------------------
// WMMA bf16 GEMM: C[M,N] = A[M,K] @ Bt[N,K]^T + bias[N]
// wave tile 32x64 (2x4 accumulators), 8 waves as 4(M) x 2(N) -> 128x128 block.
// Register double-buffering across K-steps: loads for step s+1 are issued
// before the 8 WMMAs of step s, so waits are partial, not full drains.
// Requires K % 64 == 0 (holds: K = 512 or 256 here).
// ---------------------------------------------------------------------------
__global__ __launch_bounds__(256) void wmma_gemm_bias(
    const unsigned short* __restrict__ A, const unsigned short* __restrict__ Bt,
    const float* __restrict__ bias, float* __restrict__ Cmat,
    int M, int N, int K)
{
  const int lane = threadIdx.x & 31;
  const int wave = threadIdx.x >> 5;
  const int wm = wave & 3, wn = wave >> 2;
  const int m0 = blockIdx.x * 128 + wm * 32;
  const int n0 = blockIdx.y * 128 + wn * 64;
  const int r    = lane & 15;
  const int koff = (lane >> 4) << 3;

  const __bf16* Ab = (const __bf16*)A;
  const __bf16* Bb = (const __bf16*)Bt;
  const __bf16* arow0 = Ab + (size_t)(m0 + r) * K + koff;
  const __bf16* arow1 = arow0 + (size_t)16 * K;
  const __bf16* brow  = Bb + (size_t)(n0 + r) * K + koff;

  v8f acc[2][4] = {};
  FragBF aF[2][2], bF[2][4];

  load_a2(aF[0], arow0, arow1, 0);
  load_b4(bF[0], brow, 0, K);
  const int nk = K >> 5;
  for (int s = 0; s + 2 <= nk; s += 2) {
    load_a2(aF[1], arow0, arow1, (s + 1) << 5);
    load_b4(bF[1], brow, (s + 1) << 5, K);
    wmma_step(acc, aF[0], bF[0]);
    if (s + 2 < nk) {
      load_a2(aF[0], arow0, arow1, (s + 2) << 5);
      load_b4(bF[0], brow, (s + 2) << 5, K);
    }
    wmma_step(acc, aF[1], bF[1]);
  }

  const int col = lane & 15;
  const int mhi = (lane >> 4) << 3;   // C/D layout: VGPR v -> M = v + 8*(lane>=16)
#pragma unroll
  for (int mt = 0; mt < 2; ++mt)
#pragma unroll
    for (int t = 0; t < 4; ++t) {
      const int n = n0 + t * 16 + col;
      const float bv = bias[n];
#pragma unroll
      for (int v = 0; v < 8; ++v)
        Cmat[(size_t)(m0 + mt * 16 + mhi + v) * N + n] = acc[mt][t][v] + bv;
    }
}

// ---------------------------------------------------------------------------
// Masked scatter-add of unit features into the spatial grid (f32 atomics)
// ---------------------------------------------------------------------------
__global__ __launch_bounds__(256) void scatter_add_units(
    const float* __restrict__ h, const int* __restrict__ ux,
    const int* __restrict__ uy, const unsigned char* __restrict__ mask,
    float* __restrict__ grid)
{
  const int u = blockIdx.x;
  if (!mask[u]) return;
  const int c = threadIdx.x;
  const int p = uy[u] * GX + ux[u];
  atomicAdd(&grid[(size_t)p * H2 + c], h[(size_t)u * H2 + c]);
}

// ---------------------------------------------------------------------------
// 3x3 SAME conv as 9-tap implicit WMMA GEMM over the padded halo buffer.
// wave tile: 32 pixels x 64 channels; 8 waves -> 128 pixels x 128 channels.
// Double-buffered 8-step K loop per tap (K_tap = 256).
// ---------------------------------------------------------------------------
__global__ __launch_bounds__(256) void wmma_conv3x3(
    const unsigned short* __restrict__ Gp, const unsigned short* __restrict__ Wc,
    const float* __restrict__ bias, float* __restrict__ out)
{
  const int lane = threadIdx.x & 31;
  const int wave = threadIdx.x >> 5;
  const int wm = wave & 3, wn = wave >> 2;
  const int p0 = blockIdx.x * 128 + wm * 32;   // 32 pixels, all in one grid row
  const int n0 = blockIdx.y * 128 + wn * 64;
  const int r    = lane & 15;
  const int koff = (lane >> 4) << 3;

  const __bf16* G = (const __bf16*)Gp;
  const __bf16* W = (const __bf16*)Wc;

  const int pA = p0 + r;
  const int gyA = pA >> 6, gxA = pA & 63;   // subtile 1 is gxA + 16, same row

  v8f acc[2][4] = {};
#pragma unroll
  for (int ty = 0; ty < 3; ++ty)
#pragma unroll
    for (int tx = 0; tx < 3; ++tx) {
      const __bf16* arow0 = G + ((size_t)(gyA + ty) * PAD_W + (gxA + tx)) * H2 + koff;
      const __bf16* arow1 = arow0 + (size_t)16 * H2;
      const __bf16* brow  = W + (size_t)(ty * 3 + tx) * H2 * H2
                              + (size_t)(n0 + r) * H2 + koff;
      FragBF aF[2][2], bF[2][4];
      load_a2(aF[0], arow0, arow1, 0);
      load_b4(bF[0], brow, 0, H2);
      for (int s = 0; s + 2 <= 8; s += 2) {
        load_a2(aF[1], arow0, arow1, (s + 1) << 5);
        load_b4(bF[1], brow, (s + 1) << 5, H2);
        wmma_step(acc, aF[0], bF[0]);
        if (s + 2 < 8) {
          load_a2(aF[0], arow0, arow1, (s + 2) << 5);
          load_b4(bF[0], brow, (s + 2) << 5, H2);
        }
        wmma_step(acc, aF[1], bF[1]);
      }
    }

  const int col = lane & 15;
  const int mhi = (lane >> 4) << 3;
#pragma unroll
  for (int mt = 0; mt < 2; ++mt)
#pragma unroll
    for (int t = 0; t < 4; ++t) {
      const int n = n0 + t * 16 + col;
      const float bv = bias[n];
#pragma unroll
      for (int v = 0; v < 8; ++v)
        out[(size_t)(p0 + mt * 16 + mhi + v) * H2 + n] = acc[mt][t][v] + bv;
    }
}

// ---------------------------------------------------------------------------
extern "C" void kernel_launch(void* const* d_in, const int* in_sizes, int n_in,
                              void* d_out, int out_size, void* d_ws, size_t ws_size,
                              hipStream_t stream)
{
  (void)in_sizes; (void)n_in; (void)out_size; (void)ws_size;

  const float* z      = (const float*)d_in[0];
  const int*   ux     = (const int*)d_in[1];
  const int*   uy     = (const int*)d_in[2];
  const unsigned char* mask = (const unsigned char*)d_in[3];
  const float* ln1_s  = (const float*)d_in[4];
  const float* ln1_b  = (const float*)d_in[5];
  const float* W1     = (const float*)d_in[6];
  const float* b1     = (const float*)d_in[7];
  const float* ln2_s  = (const float*)d_in[8];
  const float* ln2_b  = (const float*)d_in[9];
  const float* W2     = (const float*)d_in[10];
  const float* b2     = (const float*)d_in[11];
  const float* ln3_s  = (const float*)d_in[12];
  const float* ln3_b  = (const float*)d_in[13];
  const float* conv_w = (const float*)d_in[14];
  const float* conv_b = (const float*)d_in[15];
  float* out = (float*)d_out;

  char* ws = (char*)d_ws;
  size_t off = 0;
  auto alloc = [&](size_t bytes) -> void* {
    void* p = ws + off;
    off += (bytes + 255) & ~(size_t)255;
    return p;
  };

  unsigned short* Abf  = (unsigned short*)alloc((size_t)U_CNT * H1 * 2);   // bf16 acts (F_IN==H1)
  float*          Cbuf = (float*)alloc((size_t)U_CNT * H1 * 4);            // f32 GEMM out (reused)
  float*          grid = (float*)alloc((size_t)GY * GX * H2 * 4);
  unsigned short* Gp   = (unsigned short*)alloc((size_t)PAD_W * PAD_W * H2 * 2);
  unsigned short* W1t  = (unsigned short*)alloc((size_t)H1 * F_IN * 2);
  unsigned short* W2t  = (unsigned short*)alloc((size_t)H2 * H1 * 2);
  unsigned short* Wct  = (unsigned short*)alloc((size_t)9 * H2 * H2 * 2);

  // Weight conversions (bf16, transposed for WMMA B-fragment loading)
  cvt_transpose_bf16<<<(F_IN * H1 + 255) / 256, 256, 0, stream>>>(W1, W1t, F_IN, H1);
  cvt_transpose_bf16<<<(H1 * H2 + 255) / 256, 256, 0, stream>>>(W2, W2t, H1, H2);
  cvt_convw_bf16<<<(9 * H2 * H2 + 255) / 256, 256, 0, stream>>>(conv_w, Wct);

  // Stage 1: h = relu(LN(z)) @ W1 + b1
  ln_relu_bf16<<<U_CNT, 256, 0, stream>>>(z, ln1_s, ln1_b, Abf, F_IN);
  wmma_gemm_bias<<<dim3(U_CNT / 128, H1 / 128), 256, 0, stream>>>(
      Abf, W1t, b1, Cbuf, U_CNT, H1, F_IN);

  // Stage 2: h = relu(LN(h)) @ W2 + b2   (Cbuf reused for the [U,H2] output)
  ln_relu_bf16<<<U_CNT, 256, 0, stream>>>(Cbuf, ln2_s, ln2_b, Abf, H1);
  wmma_gemm_bias<<<dim3(U_CNT / 128, H2 / 128), 256, 0, stream>>>(
      Abf, W2t, b2, Cbuf, U_CNT, H2, H1);

  // Stage 3: masked scatter-add into grid, LN+relu into padded halo buffer
  (void)hipMemsetAsync(grid, 0, (size_t)GY * GX * H2 * sizeof(float), stream);
  (void)hipMemsetAsync(Gp, 0, (size_t)PAD_W * PAD_W * H2 * 2, stream);
  scatter_add_units<<<U_CNT, H2, 0, stream>>>(Cbuf, ux, uy, mask, grid);
  ln_relu_grid_pad<<<GY * GX, 256, 0, stream>>>(grid, ln3_s, ln3_b, Gp);

  // Stage 4: 3x3 conv via implicit WMMA GEMM
  wmma_conv3x3<<<dim3(GY * GX / 128, H2 / 128), 256, 0, stream>>>(Gp, Wct, conv_b, out);
}
